// Net_60413009985719
// MI455X (gfx1250) — matrix operand order repaired and checked by reading, not compiled, not run on hardware
//
#include <hip/hip_runtime.h>

typedef __attribute__((ext_vector_type(16))) _Float16 v16h;
typedef __attribute__((ext_vector_type(8)))  float    v8f;

#define HID 64
#define GSZ 256   // 4*HID gate rows

__device__ __forceinline__ float sigmoidf_(float x) {
    return 1.0f / (1.0f + __expf(-x));
}

// One workgroup, 8 wave32s. Computes the full network for the single relevant
// input scalar x[L-1]. Hidden matvecs use split-f16 WMMA (f32-accurate).
__launch_bounds__(256)
__global__ void lstm_head_kernel(const float* __restrict__ x, int L,
                                 const float* __restrict__ Wih0,
                                 const float* __restrict__ bih0,
                                 const float* __restrict__ bhh0,
                                 const float* __restrict__ Wih,
                                 const float* __restrict__ bih,
                                 const float* __restrict__ bhh,
                                 const float* __restrict__ fc_w,
                                 const float* __restrict__ fc_b,
                                 const float* __restrict__ mean_w,
                                 const float* __restrict__ mean_b,
                                 const float* __restrict__ ls_w,
                                 const float* __restrict__ ls_b,
                                 const float* __restrict__ c1_w,
                                 const float* __restrict__ c1_b,
                                 const float* __restrict__ c2_w,
                                 const float* __restrict__ c2_b,
                                 float* __restrict__ out)
{
    __shared__ float    gates[GSZ];
    __shared__ alignas(16) _Float16 h_hi[HID];
    __shared__ alignas(16) _Float16 h_lo[HID];
    __shared__ float    hf[HID];
    __shared__ float    zbuf[32];
    __shared__ float    ubuf[16];

    const int  tid  = threadIdx.x;
    const int  wave = tid >> 5;      // 0..7
    const int  lane = tid & 31;      // lane within wave32
    const bool hiH  = (lane >= 16);  // upper half-wave (A/B layout split)

    // ---------------- layer 0: gates = x_last * Wih0 + bih0 + bhh0 ----------
    const float xl = x[L - 1];
    gates[tid] = xl * Wih0[tid] + bih0[tid] + bhh0[tid];
    __syncthreads();

    if (tid < HID) {
        const int j = tid;
        float cv = sigmoidf_(gates[j]) * tanhf(gates[128 + j]);
        float hv = sigmoidf_(gates[192 + j]) * tanhf(cv);
        hf[j] = hv;
        _Float16 hh = (_Float16)hv;
        h_hi[j] = hh;
        h_lo[j] = (_Float16)(hv - (float)hh);
    }
    __syncthreads();

    // ---------------- layers 1..4 : gates = Wih[l] @ h + bih + bhh ----------
    for (int l = 0; l < 4; ++l) {
        const float* Wl = Wih + l * GSZ * HID;
        const float* bi = bih + l * GSZ;
        const float* bh = bhh + l * GSZ;

        // B operands (h broadcast over N columns) are tile-invariant: build
        // once per layer, keep in VGPRs across all tiles.
        // B k order chosen here (contiguous): lanes0-15 K=0..15, lanes16-31 K=16..31.
        const int kb = hiH ? 16 : 0;
        v16h bh0, bl0, bh1, bl1;
        #pragma unroll
        for (int e = 0; e < 16; ++e) {
            bh0[e] = h_hi[kb + e];           // K-chunk 0..31
            bl0[e] = h_lo[kb + e];
            bh1[e] = h_hi[32 + kb + e];      // K-chunk 32..63
            bl1[e] = h_lo[32 + kb + e];
        }

        // rows 64..127 (f gate) are dead -> 12 of 16 row-tiles
        for (int it = wave; it < 12; it += 8) {   // uniform within each wave
            const int t = it + ((it >= 4) ? 4 : 0);   // {0..3, 8..15}
            const int r = t * 16 + (lane & 15);       // A row for this lane
            const float* wr = Wl + r * HID;

            // C init = bias, per 16x16 f32 C/D layout (VGPR v: M=v / M=v+8)
            v8f c;
            #pragma unroll
            for (int v = 0; v < 8; ++v) {
                int mrow = t * 16 + (hiH ? v + 8 : v);
                c[v] = bi[mrow] + bh[mrow];
            }

            // A k order chosen here (contiguous): lanes0-15 K=0..15, lanes16-31 K=16..31 (matches B)
            const int o1 = hiH ? 16 : 0;
            const int o2 = hiH ? 24 : 8;

            #pragma unroll
            for (int cidx = 0; cidx < 2; ++cidx) {   // K-chunks 0..31, 32..63
                const int ko = cidx * 32;
                const float4* p1 = (const float4*)(wr + ko + o1);
                const float4* p2 = (const float4*)(wr + ko + o2);
                float wv[16];
                float4 q;
                q = p1[0]; wv[0] = q.x;  wv[1] = q.y;  wv[2] = q.z;  wv[3] = q.w;
                q = p1[1]; wv[4] = q.x;  wv[5] = q.y;  wv[6] = q.z;  wv[7] = q.w;
                q = p2[0]; wv[8] = q.x;  wv[9] = q.y;  wv[10] = q.z; wv[11] = q.w;
                q = p2[1]; wv[12] = q.x; wv[13] = q.y; wv[14] = q.z; wv[15] = q.w;

                v16h a_hi, a_lo;
                #pragma unroll
                for (int e = 0; e < 16; ++e) {
                    _Float16 whv = (_Float16)wv[e];
                    a_hi[e] = whv;
                    a_lo[e] = (_Float16)(wv[e] - (float)whv);
                }

                const v16h bhc = cidx ? bh1 : bh0;
                const v16h blc = cidx ? bl1 : bl0;

                // split-f16: W*h ~= Whi*hhi + Whi*hlo + Wlo*hhi  (f32 accum)
                c = __builtin_amdgcn_wmma_f32_16x16x32_f16(false, a_hi, false, bhc, (short)0, c, false, false);
                c = __builtin_amdgcn_wmma_f32_16x16x32_f16(false, a_hi, false, blc, (short)0, c, false, false);
                c = __builtin_amdgcn_wmma_f32_16x16x32_f16(false, a_lo, false, bhc, (short)0, c, false, false);
            }

            // all 16 columns of D are identical; lanes 0 and 16 spill 8 rows each
            if ((lane & 15) == 0) {
                #pragma unroll
                for (int v = 0; v < 8; ++v)
                    gates[t * 16 + (hiH ? v + 8 : v)] = c[v];
            }
        }
        __syncthreads();

        if (tid < HID) {
            const int j = tid;
            float cv = sigmoidf_(gates[j]) * tanhf(gates[128 + j]);
            float hv = sigmoidf_(gates[192 + j]) * tanhf(cv);
            hf[j] = hv;
            _Float16 hh = (_Float16)hv;
            h_hi[j] = hh;
            h_lo[j] = (_Float16)(hv - (float)hh);
        }
        __syncthreads();
    }

    // ---------------- head (tiny, exact f32 VALU) ----------------
    if (tid < 32) {
        float acc = fc_b[tid];
        const float* fw = fc_w + tid * HID;
        #pragma unroll 8
        for (int k = 0; k < HID; ++k) acc = fmaf(fw[k], hf[k], acc);
        zbuf[tid] = fmaxf(acc, 0.0f);
    }
    __syncthreads();
    if (tid < 16) {
        float a2 = c1_b[tid];
        const float* cw = c1_w + tid * 32;
        #pragma unroll 8
        for (int k = 0; k < 32; ++k) a2 = fmaf(cw[k], zbuf[k], a2);
        ubuf[tid] = fmaxf(a2, 0.0f);
    }
    __syncthreads();
    if (tid == 0) {
        float m = mean_b[0], s = ls_b[0];
        #pragma unroll 8
        for (int k = 0; k < 32; ++k) {
            m = fmaf(mean_w[k], zbuf[k], m);
            s = fmaf(ls_w[k], zbuf[k], s);
        }
        float vv = c2_b[0];
        #pragma unroll 8
        for (int k = 0; k < 16; ++k) vv = fmaf(c2_w[k], ubuf[k], vv);
        out[0] = m;   // mean
        out[1] = s;   // log_std
        out[2] = vv;  // v
    }
}

extern "C" void kernel_launch(void* const* d_in, const int* in_sizes, int n_in,
                              void* d_out, int out_size, void* d_ws, size_t ws_size,
                              hipStream_t stream) {
    (void)n_in; (void)out_size; (void)d_ws; (void)ws_size;
    const float* x      = (const float*)d_in[0];
    const float* Wih0   = (const float*)d_in[1];
    // d_in[2] = Whh0 : dead (h_prev == 0)
    const float* bih0   = (const float*)d_in[3];
    const float* bhh0   = (const float*)d_in[4];
    const float* Wih    = (const float*)d_in[5];
    // d_in[6] = Whh : dead
    const float* bih    = (const float*)d_in[7];
    const float* bhh    = (const float*)d_in[8];
    const float* fc_w   = (const float*)d_in[9];
    const float* fc_b   = (const float*)d_in[10];
    const float* mean_w = (const float*)d_in[11];
    const float* mean_b = (const float*)d_in[12];
    const float* ls_w   = (const float*)d_in[13];
    const float* ls_b   = (const float*)d_in[14];
    const float* c1_w   = (const float*)d_in[15];
    const float* c1_b   = (const float*)d_in[16];
    const float* c2_w   = (const float*)d_in[17];
    const float* c2_b   = (const float*)d_in[18];
    const int L = in_sizes[0];

    lstm_head_kernel<<<1, 256, 0, stream>>>(
        x, L, Wih0, bih0, bhh0, Wih, bih, bhh,
        fc_w, fc_b, mean_w, mean_b, ls_w, ls_b,
        c1_w, c1_b, c2_w, c2_b, (float*)d_out);
}
